// SwitchingNRIModel_73461120631441
// MI455X (gfx1250) — compile-verified, matrix operand order checked by reading, not run on hardware
//
#include <hip/hip_runtime.h>
#include <hip/hip_bf16.h>

typedef __attribute__((ext_vector_type(16))) _Float16 v16h;
typedef __attribute__((ext_vector_type(8)))  float    v8f;
typedef _Float16 half_t;

#define DEV __device__ __forceinline__

static constexpr int Nn  = 4000;
static constexpr int Ee  = 64000;
static constexpr int Tt  = 4;
static constexpr int NT  = Nn * Tt;        // 16000
static constexpr int ET  = Ee * Tt;        // 256000
static constexpr int E3  = Ee * 3;         // 192000
static constexpr int N3  = Nn * 3;         // 12000

DEV v8f wmma16(v16h a, v16h b, v8f c) {
  return __builtin_amdgcn_wmma_f32_16x16x32_f16(false, a, false, b, (short)0, c, false, false);
}
DEV float relu_f(float x) { return x > 0.f ? x : 0.f; }
DEV float sigmoid_f(float x) { return 1.f / (1.f + __expf(-x)); }
DEV float tanh_f(float x) { float e = __expf(2.f * x); return (e - 1.f) / (e + 1.f); }

// ---------------------------------------------------------------- utilities
__global__ void zero_f32_kernel(float* p, long long n) {
  long long i = (long long)blockIdx.x * blockDim.x + threadIdx.x;
  long long stride = (long long)gridDim.x * blockDim.x;
  for (; i < n; i += stride) p[i] = 0.f;
}

// dst[n*Kpad+k] = k<K ? (f16)src[k*N+n] : 0   (weight transpose to B-layout)
__global__ void transpose_w_kernel(const float* __restrict__ src, half_t* __restrict__ dst,
                                   int K, int N, int Kpad) {
  int idx = blockIdx.x * blockDim.x + threadIdx.x;
  if (idx >= N * Kpad) return;
  int n = idx / Kpad, k = idx - n * Kpad;
  dst[idx] = (k < K) ? (half_t)src[k * N + n] : (half_t)0.f;
}

__global__ void cvt_f32_to_f16_kernel(const float* __restrict__ src, half_t* __restrict__ dst, int n) {
  int i = blockIdx.x * blockDim.x + threadIdx.x;
  if (i < n) dst[i] = (half_t)src[i];
}

// ---------------------------------------------------------------- projections (K=4, VALU)
__global__ void proj_kernel(const float* __restrict__ in, const float* __restrict__ W,
                            const float* __restrict__ b, half_t* __restrict__ out, int rows) {
  int idx = blockIdx.x * blockDim.x + threadIdx.x;
  if (idx >= rows * 128) return;
  int r = idx >> 7, c = idx & 127;
  const float* ir = in + (size_t)r * 4;
  float s = b[c] + ir[0]*W[c] + ir[1]*W[128+c] + ir[2]*W[256+c] + ir[3]*W[384+c];
  out[idx] = (half_t)relu_f(s);
}

// ---------------------------------------------------------------- MPNN edge MLP (WMMA)
// rows = E*T; A = [x[tgt] | x[src] | e]  (K=384) -> relu -> K=128 -> out (no act)
__global__ __launch_bounds__(256) void edge_mlp_kernel(
    const half_t* __restrict__ xf, const half_t* __restrict__ ef,
    const int* __restrict__ src, const int* __restrict__ tgt,
    const half_t* __restrict__ W1T, const float* __restrict__ b1,
    const half_t* __restrict__ W2T, const float* __restrict__ b2,
    half_t* __restrict__ out) {
  __shared__ __align__(32) half_t Hs[8][16 * 128];
  __builtin_prefetch((const void*)W1T, 0, 1);
  __builtin_prefetch((const void*)W2T, 0, 1);
  int wave = threadIdx.x >> 5, lane = threadIdx.x & 31;
  int tile = blockIdx.x * 8 + wave;
  int lrow = lane & 15, hl = lane >> 4;
  int grow = tile * 16 + lrow;
  int e = grow >> 2, t = grow & 3;
  int si = src[e], ti = tgt[e];
  const half_t* seg0 = xf + (size_t)(ti * 4 + t) * 128;
  const half_t* seg1 = xf + (size_t)(si * 4 + t) * 128;
  const half_t* seg2 = ef + (size_t)grow * 128;

  half_t* hs = Hs[wave];
  // ---- layer 1: hold all 8 N-tile accumulators, batch B-fragment loads
  v8f acc[8];
#pragma unroll
  for (int nt = 0; nt < 8; ++nt) acc[nt] = (v8f){};
#pragma unroll
  for (int s = 0; s < 12; ++s) {
    const half_t* base = (s < 4) ? seg0 : (s < 8) ? seg1 : seg2;
    v16h a = *(const v16h*)(base + (s & 3) * 32 + hl * 16);
    v16h bq[4];
#pragma unroll
    for (int q = 0; q < 4; ++q)
      bq[q] = *(const v16h*)(W1T + (size_t)(q * 16 + lrow) * 384 + s * 32 + hl * 16);
#pragma unroll
    for (int q = 0; q < 4; ++q) acc[q] = wmma16(a, bq[q], acc[q]);
#pragma unroll
    for (int q = 0; q < 4; ++q)
      bq[q] = *(const v16h*)(W1T + (size_t)((q + 4) * 16 + lrow) * 384 + s * 32 + hl * 16);
#pragma unroll
    for (int q = 0; q < 4; ++q) acc[q + 4] = wmma16(a, bq[q], acc[q + 4]);
  }
#pragma unroll
  for (int nt = 0; nt < 8; ++nt) {
    float bias = b1[nt * 16 + lrow];
#pragma unroll
    for (int v = 0; v < 8; ++v)
      hs[(v + 8 * hl) * 128 + nt * 16 + lrow] = (half_t)relu_f(acc[nt][v] + bias);
  }
  asm volatile("s_wait_dscnt 0" ::: "memory");
  // ---- layer 2
#pragma unroll
  for (int nt = 0; nt < 8; ++nt) acc[nt] = (v8f){};
#pragma unroll
  for (int s = 0; s < 4; ++s) {
    v16h a = *(const v16h*)(hs + lrow * 128 + s * 32 + hl * 16);
    v16h bq[4];
#pragma unroll
    for (int q = 0; q < 4; ++q)
      bq[q] = *(const v16h*)(W2T + (size_t)(q * 16 + lrow) * 128 + s * 32 + hl * 16);
#pragma unroll
    for (int q = 0; q < 4; ++q) acc[q] = wmma16(a, bq[q], acc[q]);
#pragma unroll
    for (int q = 0; q < 4; ++q)
      bq[q] = *(const v16h*)(W2T + (size_t)((q + 4) * 16 + lrow) * 128 + s * 32 + hl * 16);
#pragma unroll
    for (int q = 0; q < 4; ++q) acc[q + 4] = wmma16(a, bq[q], acc[q + 4]);
  }
#pragma unroll
  for (int nt = 0; nt < 8; ++nt) {
    float bias = b2[nt * 16 + lrow];
#pragma unroll
    for (int v = 0; v < 8; ++v)
      out[(size_t)(tile * 16 + v + 8 * hl) * 128 + nt * 16 + lrow] = (half_t)(acc[nt][v] + bias);
  }
}

// ---------------------------------------------------------------- MPNN node MLP + segment sum
// A = [x[tgt] | e2] (K=256) -> relu -> K=128 -> atomicAdd into x2[tgt_row]
__global__ __launch_bounds__(256) void node_mlp_kernel(
    const half_t* __restrict__ xf, const half_t* __restrict__ e2,
    const int* __restrict__ tgt,
    const half_t* __restrict__ W1T, const float* __restrict__ b1,
    const half_t* __restrict__ W2T, const float* __restrict__ b2,
    float* __restrict__ x2) {
  __shared__ __align__(32) half_t Hs[8][16 * 128];
  __builtin_prefetch((const void*)W1T, 0, 1);
  __builtin_prefetch((const void*)W2T, 0, 1);
  int wave = threadIdx.x >> 5, lane = threadIdx.x & 31;
  int tile = blockIdx.x * 8 + wave;
  int lrow = lane & 15, hl = lane >> 4;
  int grow = tile * 16 + lrow;
  int e = grow >> 2, t = grow & 3;
  int ti = tgt[e];
  const half_t* seg0 = xf + (size_t)(ti * 4 + t) * 128;
  const half_t* seg1 = e2 + (size_t)grow * 128;

  half_t* hs = Hs[wave];
  v8f acc[8];
#pragma unroll
  for (int nt = 0; nt < 8; ++nt) acc[nt] = (v8f){};
#pragma unroll
  for (int s = 0; s < 8; ++s) {
    const half_t* base = (s < 4) ? seg0 : seg1;
    v16h a = *(const v16h*)(base + (s & 3) * 32 + hl * 16);
    v16h bq[4];
#pragma unroll
    for (int q = 0; q < 4; ++q)
      bq[q] = *(const v16h*)(W1T + (size_t)(q * 16 + lrow) * 256 + s * 32 + hl * 16);
#pragma unroll
    for (int q = 0; q < 4; ++q) acc[q] = wmma16(a, bq[q], acc[q]);
#pragma unroll
    for (int q = 0; q < 4; ++q)
      bq[q] = *(const v16h*)(W1T + (size_t)((q + 4) * 16 + lrow) * 256 + s * 32 + hl * 16);
#pragma unroll
    for (int q = 0; q < 4; ++q) acc[q + 4] = wmma16(a, bq[q], acc[q + 4]);
  }
#pragma unroll
  for (int nt = 0; nt < 8; ++nt) {
    float bias = b1[nt * 16 + lrow];
#pragma unroll
    for (int v = 0; v < 8; ++v)
      hs[(v + 8 * hl) * 128 + nt * 16 + lrow] = (half_t)relu_f(acc[nt][v] + bias);
  }
  asm volatile("s_wait_dscnt 0" ::: "memory");

  int trows[8];
#pragma unroll
  for (int v = 0; v < 8; ++v) {
    int gm = tile * 16 + v + 8 * hl;
    int em = gm >> 2, tm = gm & 3;
    trows[v] = tgt[em] * 4 + tm;
  }
#pragma unroll
  for (int nt = 0; nt < 8; ++nt) acc[nt] = (v8f){};
#pragma unroll
  for (int s = 0; s < 4; ++s) {
    v16h a = *(const v16h*)(hs + lrow * 128 + s * 32 + hl * 16);
    v16h bq[4];
#pragma unroll
    for (int q = 0; q < 4; ++q)
      bq[q] = *(const v16h*)(W2T + (size_t)(q * 16 + lrow) * 128 + s * 32 + hl * 16);
#pragma unroll
    for (int q = 0; q < 4; ++q) acc[q] = wmma16(a, bq[q], acc[q]);
#pragma unroll
    for (int q = 0; q < 4; ++q)
      bq[q] = *(const v16h*)(W2T + (size_t)((q + 4) * 16 + lrow) * 128 + s * 32 + hl * 16);
#pragma unroll
    for (int q = 0; q < 4; ++q) acc[q + 4] = wmma16(a, bq[q], acc[q + 4]);
  }
#pragma unroll
  for (int nt = 0; nt < 8; ++nt) {
    float bias = b2[nt * 16 + lrow];
#pragma unroll
    for (int v = 0; v < 8; ++v)
      unsafeAtomicAdd(&x2[(size_t)trows[v] * 128 + nt * 16 + lrow], acc[nt][v] + bias);
  }
}

// ---------------------------------------------------------------- logit head (N=4, VALU)
__global__ void raw_kernel(const half_t* __restrict__ e2, const float* __restrict__ W,
                           const float* __restrict__ b, float* __restrict__ raw, int rows) {
  int idx = blockIdx.x * blockDim.x + threadIdx.x;
  if (idx >= rows * 4) return;
  int r = idx >> 2, k = idx & 3;
  const half_t* er = e2 + (size_t)r * 128;
  float s = b[k];
  for (int j = 0; j < 128; ++j) s += (float)er[j] * W[j * 4 + k];
  raw[idx] = s;
}

// ---------------------------------------------------------------- bidirectional GRU (WMMA h@Whh)
__global__ __launch_bounds__(128) void gru_kernel(
    const float* __restrict__ raw,
    const half_t* __restrict__ WhhT_f, const float* __restrict__ Wih_f,
    const float* __restrict__ bih_f, const float* __restrict__ bhh_f,
    const half_t* __restrict__ WhhT_b, const float* __restrict__ Wih_b,
    const float* __restrict__ bih_b, const float* __restrict__ bhh_b,
    half_t* __restrict__ hf, half_t* __restrict__ hb) {
  __shared__ __align__(32) half_t Hbuf[4][2][16 * 128];
  __shared__ float rawS[4][256];
  int wave = threadIdx.x >> 5, lane = threadIdx.x & 31;
  int task = blockIdx.x * 4 + wave;
  int tile = task >> 1, dir = task & 1;
  int eBase = tile * 16;
  int lrow = lane & 15, hl = lane >> 4;
  const half_t* WhhT = dir ? WhhT_b : WhhT_f;
  const float*  Wih  = dir ? Wih_b  : Wih_f;
  const float*  bih  = dir ? bih_b  : bih_f;
  const float*  bhh  = dir ? bhh_b  : bhh_f;
  half_t* yout = dir ? hb : hf;
  __builtin_prefetch((const void*)WhhT, 0, 1);

  float* rs = rawS[wave];
#pragma unroll
  for (int i = 0; i < 8; ++i) rs[lane * 8 + i] = raw[(size_t)eBase * 16 + lane * 8 + i];
  uint32_t* hz = (uint32_t*)Hbuf[wave][0];
#pragma unroll
  for (int i = 0; i < 32; ++i) hz[lane * 32 + i] = 0u;
  asm volatile("s_wait_dscnt 0" ::: "memory");

  int cur = 0;
  for (int ts = 0; ts < 4; ++ts) {
    int t = dir ? 3 - ts : ts;
    half_t* hA = Hbuf[wave][cur];
    half_t* hB = Hbuf[wave][cur ^ 1];
    v16h A[4];
#pragma unroll
    for (int s = 0; s < 4; ++s) A[s] = *(const v16h*)(hA + lrow * 128 + s * 32 + hl * 16);
#pragma unroll
    for (int nt = 0; nt < 8; ++nt) {
      int j = nt * 16 + lrow;
      v8f cr = {}, cz = {}, cn = {};
#pragma unroll
      for (int s = 0; s < 4; ++s) {
        v16h bg[3];
        bg[0] = *(const v16h*)(WhhT + (size_t)(      j) * 128 + s * 32 + hl * 16);
        bg[1] = *(const v16h*)(WhhT + (size_t)(128 + j) * 128 + s * 32 + hl * 16);
        bg[2] = *(const v16h*)(WhhT + (size_t)(256 + j) * 128 + s * 32 + hl * 16);
        cr = wmma16(A[s], bg[0], cr);
        cz = wmma16(A[s], bg[1], cz);
        cn = wmma16(A[s], bg[2], cn);
      }
      float wr0 = Wih[j],       wr1 = Wih[384 + j],  wr2 = Wih[768 + j],  wr3 = Wih[1152 + j];
      float wz0 = Wih[128 + j], wz1 = Wih[512 + j],  wz2 = Wih[896 + j],  wz3 = Wih[1280 + j];
      float wn0 = Wih[256 + j], wn1 = Wih[640 + j],  wn2 = Wih[1024 + j], wn3 = Wih[1408 + j];
      float bir = bih[j], biz = bih[128 + j], bin = bih[256 + j];
      float bhr = bhh[j], bhz = bhh[128 + j], bhn = bhh[256 + j];
#pragma unroll
      for (int v = 0; v < 8; ++v) {
        int m = v + 8 * hl;
        float x0 = rs[m * 16 + t * 4 + 0], x1 = rs[m * 16 + t * 4 + 1];
        float x2 = rs[m * 16 + t * 4 + 2], x3 = rs[m * 16 + t * 4 + 3];
        float gir = bir + x0 * wr0 + x1 * wr1 + x2 * wr2 + x3 * wr3;
        float giz = biz + x0 * wz0 + x1 * wz1 + x2 * wz2 + x3 * wz3;
        float gin = bin + x0 * wn0 + x1 * wn1 + x2 * wn2 + x3 * wn3;
        float r = sigmoid_f(gir + cr[v] + bhr);
        float z = sigmoid_f(giz + cz[v] + bhz);
        float nn = tanh_f(gin + r * (cn[v] + bhn));
        float hp = (float)hA[m * 128 + j];
        float h2 = (1.f - z) * nn + z * hp;
        hB[m * 128 + j] = (half_t)h2;
        yout[(size_t)((eBase + m) * 4 + t) * 128 + j] = (half_t)h2;
      }
    }
    cur ^= 1;
    asm volatile("s_wait_dscnt 0" ::: "memory");
  }
}

// ---------------------------------------------------------------- gru_out + gumbel softmax
__global__ void probs_kernel(const half_t* __restrict__ hf, const half_t* __restrict__ hb,
                             const float* __restrict__ Wg, const float* __restrict__ bg,
                             const float* __restrict__ gumbel, float* __restrict__ probs, int rows) {
  int r = blockIdx.x * blockDim.x + threadIdx.x;
  if (r >= rows) return;
  float lg[4] = {bg[0], bg[1], bg[2], bg[3]};
  const half_t* f = hf + (size_t)r * 128;
  const half_t* b = hb + (size_t)r * 128;
  for (int j = 0; j < 128; ++j) {
    float v = (float)f[j];
    lg[0] += v * Wg[j * 4 + 0]; lg[1] += v * Wg[j * 4 + 1];
    lg[2] += v * Wg[j * 4 + 2]; lg[3] += v * Wg[j * 4 + 3];
  }
  for (int j = 0; j < 128; ++j) {
    float v = (float)b[j];
    lg[0] += v * Wg[(128 + j) * 4 + 0]; lg[1] += v * Wg[(128 + j) * 4 + 1];
    lg[2] += v * Wg[(128 + j) * 4 + 2]; lg[3] += v * Wg[(128 + j) * 4 + 3];
  }
  float mx = -1e30f;
#pragma unroll
  for (int k = 0; k < 4; ++k) { lg[k] = (lg[k] + gumbel[(size_t)r * 4 + k]) * 2.0f; mx = fmaxf(mx, lg[k]); }
  float sum = 0.f;
#pragma unroll
  for (int k = 0; k < 4; ++k) { lg[k] = __expf(lg[k] - mx); sum += lg[k]; }
  float inv = 1.f / sum;
#pragma unroll
  for (int k = 0; k < 4; ++k) probs[(size_t)r * 4 + k] = lg[k] * inv;
}

// ---------------------------------------------------------------- decoder message MLPs + segment sum
__global__ __launch_bounds__(256) void dec_msg_kernel(
    const float* __restrict__ x, const int* __restrict__ src, const int* __restrict__ tgt,
    const float* __restrict__ probs,
    const half_t* __restrict__ decW1T, const float* __restrict__ decB1,
    const half_t* __restrict__ decW2T, const float* __restrict__ decB2,
    float* __restrict__ aggr) {
  __shared__ __align__(32) half_t Hs[8][16 * 128];
  __shared__ float pcS[8][64];
  __builtin_prefetch((const void*)decW2T, 0, 1);
  int wave = threadIdx.x >> 5, lane = threadIdx.x & 31;
  int tile = blockIdx.x * 8 + wave;
  int lrow = lane & 15, hl = lane >> 4;
  int grow = tile * 16 + lrow;
  int e = grow / 3, t = grow - e * 3;
  int si = src[e], ti = tgt[e];

  v16h a1 = {};
  if (hl == 0) {
    const float* pi = x + (size_t)ti * 16 + t * 4;
    const float* pj = x + (size_t)si * 16 + t * 4;
#pragma unroll
    for (int q = 0; q < 4; ++q) { a1[q] = (half_t)pi[q]; a1[4 + q] = (half_t)pj[q]; }
  }
  float* pcs = pcS[wave];
#pragma unroll
  for (int i = 0; i < 2; ++i) {
    int idx = lane * 2 + i;
    int m = idx >> 2, k = idx & 3;
    int gm = tile * 16 + m;
    int em = gm / 3, tm = gm - em * 3;
    pcs[idx] = probs[(size_t)(em * 4 + tm) * 4 + k];
  }
  asm volatile("s_wait_dscnt 0" ::: "memory");

  float macc[8][8];
#pragma unroll
  for (int nt = 0; nt < 8; ++nt)
#pragma unroll
    for (int v = 0; v < 8; ++v) macc[nt][v] = 0.f;

  half_t* hs = Hs[wave];
  for (int k = 0; k < 4; ++k) {
    const half_t* W1k = decW1T + (size_t)k * 128 * 32;
    const half_t* W2k = decW2T + (size_t)k * 128 * 128;
    const float*  b1k = decB1 + k * 128;
    const float*  b2k = decB2 + k * 128;
    // layer 1: single K-step (K=8 zero-padded to 32), batch 4+4 B fragments
    v8f acc[8];
    {
      v16h bq[4];
#pragma unroll
      for (int q = 0; q < 4; ++q)
        bq[q] = *(const v16h*)(W1k + (size_t)(q * 16 + lrow) * 32 + hl * 16);
#pragma unroll
      for (int q = 0; q < 4; ++q) { v8f cz = {}; acc[q] = wmma16(a1, bq[q], cz); }
#pragma unroll
      for (int q = 0; q < 4; ++q)
        bq[q] = *(const v16h*)(W1k + (size_t)((q + 4) * 16 + lrow) * 32 + hl * 16);
#pragma unroll
      for (int q = 0; q < 4; ++q) { v8f cz = {}; acc[q + 4] = wmma16(a1, bq[q], cz); }
    }
#pragma unroll
    for (int nt = 0; nt < 8; ++nt) {
      float bias = b1k[nt * 16 + lrow];
#pragma unroll
      for (int v = 0; v < 8; ++v)
        hs[(v + 8 * hl) * 128 + nt * 16 + lrow] = (half_t)relu_f(acc[nt][v] + bias);
    }
    asm volatile("s_wait_dscnt 0" ::: "memory");
    // layer 2
#pragma unroll
    for (int nt = 0; nt < 8; ++nt) acc[nt] = (v8f){};
#pragma unroll
    for (int s = 0; s < 4; ++s) {
      v16h a = *(const v16h*)(hs + lrow * 128 + s * 32 + hl * 16);
      v16h bq[4];
#pragma unroll
      for (int q = 0; q < 4; ++q)
        bq[q] = *(const v16h*)(W2k + (size_t)(q * 16 + lrow) * 128 + s * 32 + hl * 16);
#pragma unroll
      for (int q = 0; q < 4; ++q) acc[q] = wmma16(a, bq[q], acc[q]);
#pragma unroll
      for (int q = 0; q < 4; ++q)
        bq[q] = *(const v16h*)(W2k + (size_t)((q + 4) * 16 + lrow) * 128 + s * 32 + hl * 16);
#pragma unroll
      for (int q = 0; q < 4; ++q) acc[q + 4] = wmma16(a, bq[q], acc[q + 4]);
    }
#pragma unroll
    for (int nt = 0; nt < 8; ++nt) {
      float bias = b2k[nt * 16 + lrow];
#pragma unroll
      for (int v = 0; v < 8; ++v)
        macc[nt][v] += relu_f(acc[nt][v] + bias) * pcs[(v + 8 * hl) * 4 + k];
    }
  }
  int trows[8];
#pragma unroll
  for (int v = 0; v < 8; ++v) {
    int gm = tile * 16 + v + 8 * hl;
    int em = gm / 3, tm = gm - em * 3;
    trows[v] = tgt[em] * 3 + tm;
  }
#pragma unroll
  for (int nt = 0; nt < 8; ++nt)
#pragma unroll
    for (int v = 0; v < 8; ++v)
      unsafeAtomicAdd(&aggr[(size_t)trows[v] * 128 + nt * 16 + lrow], macc[nt][v]);
}

// ---------------------------------------------------------------- decoder output head (12k rows, VALU)
__global__ __launch_bounds__(128) void dec_out_kernel(
    const float* __restrict__ x, const float* __restrict__ aggr,
    const float* __restrict__ W1, const float* __restrict__ b1,
    const float* __restrict__ W2, const float* __restrict__ b2,
    const float* __restrict__ W3, const float* __restrict__ b3,
    float* __restrict__ out) {
  __shared__ float h0[132], h1s[128], h2s[128];
  int row = blockIdx.x, tid = threadIdx.x;
  int n = row / 3, t = row - n * 3;
  if (tid < 4) h0[tid] = x[(size_t)n * 16 + t * 4 + tid];
  h0[4 + tid] = aggr[(size_t)row * 128 + tid];
  __syncthreads();
  float s = b1[tid];
  for (int i = 0; i < 132; ++i) s += h0[i] * W1[i * 128 + tid];
  h1s[tid] = relu_f(s);
  __syncthreads();
  s = b2[tid];
  for (int i = 0; i < 128; ++i) s += h1s[i] * W2[i * 128 + tid];
  h2s[tid] = relu_f(s);
  __syncthreads();
  if (tid < 4) {
    float d = b3[tid];
    for (int j = 0; j < 128; ++j) d += h2s[j] * W3[j * 4 + tid];
    out[(size_t)row * 4 + tid] = h0[tid] + d;
  }
}

// ================================================================ host launch
extern "C" void kernel_launch(void* const* d_in, const int* in_sizes, int n_in,
                              void* d_out, int out_size, void* d_ws, size_t ws_size,
                              hipStream_t stream) {
  const float* x        = (const float*)d_in[0];
  const float* edge_att = (const float*)d_in[1];
  const int*   eidx     = (const int*)d_in[2];
  const float* gumbel   = (const float*)d_in[3];
  const int* src = eidx, *tgt = eidx + Ee;

  // params (setup_inputs dict insertion order, tuples flattened in order)
  const float* npW  = (const float*)d_in[4];   const float* npB  = (const float*)d_in[5];
  const float* epW  = (const float*)d_in[6];   const float* epB  = (const float*)d_in[7];
  const float* e1W1 = (const float*)d_in[8];   const float* e1B1 = (const float*)d_in[9];
  const float* e1W2 = (const float*)d_in[10];  const float* e1B2 = (const float*)d_in[11];
  const float* n1W1 = (const float*)d_in[12];  const float* n1B1 = (const float*)d_in[13];
  const float* n1W2 = (const float*)d_in[14];  const float* n1B2 = (const float*)d_in[15];
  const float* e2W1 = (const float*)d_in[16];  const float* e2B1 = (const float*)d_in[17];
  const float* e2W2 = (const float*)d_in[18];  const float* e2B2 = (const float*)d_in[19];
  // mpnn2 node weights (20..23) are dead: xe from mpnn2 is never consumed
  const float* lhW  = (const float*)d_in[24];  const float* lhB  = (const float*)d_in[25];
  const float* gfWih = (const float*)d_in[26]; const float* gfWhh = (const float*)d_in[27];
  const float* gfBih = (const float*)d_in[28]; const float* gfBhh = (const float*)d_in[29];
  const float* gbWih = (const float*)d_in[30]; const float* gbWhh = (const float*)d_in[31];
  const float* gbBih = (const float*)d_in[32]; const float* gbBhh = (const float*)d_in[33];
  const float* goW  = (const float*)d_in[34];  const float* goB  = (const float*)d_in[35];
  const float* doW1 = (const float*)d_in[52];  const float* doB1 = (const float*)d_in[53];
  const float* doW2 = (const float*)d_in[54];  const float* doB2 = (const float*)d_in[55];
  const float* doW3 = (const float*)d_in[56];  const float* doB3 = (const float*)d_in[57];

  // workspace layout
  char* ws = (char*)d_ws;
  size_t off = 0;
  auto alloc = [&](size_t bytes) { size_t o = off; off = (off + bytes + 255) & ~(size_t)255; return o; };
  half_t* xin  = (half_t*)(ws + alloc((size_t)NT * 128 * 2));
  half_t* x1b  = (half_t*)(ws + alloc((size_t)NT * 128 * 2));
  half_t* ein  = (half_t*)(ws + alloc((size_t)ET * 128 * 2));
  half_t* e2a  = (half_t*)(ws + alloc((size_t)ET * 128 * 2));
  half_t* e2b  = (half_t*)(ws + alloc((size_t)ET * 128 * 2));
  half_t* hf   = (half_t*)(ws + alloc((size_t)ET * 128 * 2));
  half_t* hb   = (half_t*)(ws + alloc((size_t)ET * 128 * 2));
  float*  x2   = (float*)(ws + alloc((size_t)NT * 128 * 4));
  float*  raw  = (float*)(ws + alloc((size_t)ET * 4 * 4));
  float*  prb  = (float*)(ws + alloc((size_t)ET * 4 * 4));
  float*  aggr = (float*)(ws + alloc((size_t)N3 * 128 * 4));
  half_t* W1eT1 = (half_t*)(ws + alloc(128 * 384 * 2));
  half_t* W2eT1 = (half_t*)(ws + alloc(128 * 128 * 2));
  half_t* W1eT2 = (half_t*)(ws + alloc(128 * 384 * 2));
  half_t* W2eT2 = (half_t*)(ws + alloc(128 * 128 * 2));
  half_t* Wn1T  = (half_t*)(ws + alloc(128 * 256 * 2));
  half_t* Wn2T  = (half_t*)(ws + alloc(128 * 128 * 2));
  half_t* WhhTf = (half_t*)(ws + alloc(384 * 128 * 2));
  half_t* WhhTb = (half_t*)(ws + alloc(384 * 128 * 2));
  half_t* dW1T  = (half_t*)(ws + alloc(4 * 128 * 32 * 2));
  half_t* dW2T  = (half_t*)(ws + alloc(4 * 128 * 128 * 2));
  float*  dB1   = (float*)(ws + alloc(4 * 128 * 4));
  float*  dB2   = (float*)(ws + alloc(4 * 128 * 4));

  auto tw = [&](const float* s, half_t* d, int K, int N, int Kpad) {
    int tot = N * Kpad;
    hipLaunchKernelGGL(transpose_w_kernel, dim3((tot + 255) / 256), dim3(256), 0, stream, s, d, K, N, Kpad);
  };

  // zero accumulators
  hipLaunchKernelGGL(zero_f32_kernel, dim3(512), dim3(256), 0, stream, x2, (long long)NT * 128);
  hipLaunchKernelGGL(zero_f32_kernel, dim3(512), dim3(256), 0, stream, aggr, (long long)N3 * 128);

  // weight prep: transpose to N-major f16 (B-fragment layout)
  tw(e1W1, W1eT1, 384, 128, 384);  tw(e1W2, W2eT1, 128, 128, 128);
  tw(e2W1, W1eT2, 384, 128, 384);  tw(e2W2, W2eT2, 128, 128, 128);
  tw(n1W1, Wn1T, 256, 128, 256);   tw(n1W2, Wn2T, 128, 128, 128);
  tw(gfWhh, WhhTf, 128, 384, 128); tw(gbWhh, WhhTb, 128, 384, 128);
  for (int k = 0; k < 4; ++k) {
    tw((const float*)d_in[36 + 4 * k], dW1T + (size_t)k * 128 * 32, 8, 128, 32);
    tw((const float*)d_in[38 + 4 * k], dW2T + (size_t)k * 128 * 128, 128, 128, 128);
    hipMemcpyAsync(dB1 + k * 128, d_in[37 + 4 * k], 128 * 4, hipMemcpyDeviceToDevice, stream);
    hipMemcpyAsync(dB2 + k * 128, d_in[39 + 4 * k], 128 * 4, hipMemcpyDeviceToDevice, stream);
  }

  // encoder projections
  hipLaunchKernelGGL(proj_kernel, dim3(NT * 128 / 256), dim3(256), 0, stream, x, npW, npB, xin, NT);
  hipLaunchKernelGGL(proj_kernel, dim3(ET * 128 / 256), dim3(256), 0, stream, edge_att, epW, epB, ein, ET);

  // mpnn1
  hipLaunchKernelGGL(edge_mlp_kernel, dim3(ET / 16 / 8), dim3(256), 0, stream,
                     xin, ein, src, tgt, W1eT1, e1B1, W2eT1, e1B2, e2a);
  hipLaunchKernelGGL(node_mlp_kernel, dim3(ET / 16 / 8), dim3(256), 0, stream,
                     xin, e2a, tgt, Wn1T, n1B1, Wn2T, n1B2, x2);
  hipLaunchKernelGGL(cvt_f32_to_f16_kernel, dim3(NT * 128 / 256), dim3(256), 0, stream,
                     x2, x1b, NT * 128);

  // mpnn2 (edge branch only; node branch is dead)
  hipLaunchKernelGGL(edge_mlp_kernel, dim3(ET / 16 / 8), dim3(256), 0, stream,
                     x1b, e2a, src, tgt, W1eT2, e2B1, W2eT2, e2B2, e2b);

  // logit head -> raw
  hipLaunchKernelGGL(raw_kernel, dim3(ET * 4 / 256), dim3(256), 0, stream, e2b, lhW, lhB, raw, ET);

  // bidirectional GRU
  hipLaunchKernelGGL(gru_kernel, dim3((Ee / 16) * 2 / 4), dim3(128), 0, stream,
                     raw, WhhTf, gfWih, gfBih, gfBhh, WhhTb, gbWih, gbBih, gbBhh, hf, hb);

  // gru_out + gumbel softmax
  hipLaunchKernelGGL(probs_kernel, dim3(ET / 256), dim3(256), 0, stream,
                     hf, hb, goW, goB, gumbel, prb, ET);

  // decoder message MLPs + segment sum
  hipLaunchKernelGGL(dec_msg_kernel, dim3(E3 / 16 / 8), dim3(256), 0, stream,
                     x, src, tgt, prb, dW1T, dB1, dW2T, dB2, aggr);

  // decoder output head
  hipLaunchKernelGGL(dec_out_kernel, dim3(N3), dim3(128), 0, stream,
                     x, aggr, doW1, doB1, doW2, doB2, doW3, doB3, (float*)d_out);
}
